// DynamicGRU_15796889715501
// MI455X (gfx1250) — matrix operand order repaired and checked by reading, compiled, not run on hardware
//
#include <hip/hip_runtime.h>

typedef _Float16 half_t;
typedef __attribute__((ext_vector_type(16))) _Float16 v16h;
typedef __attribute__((ext_vector_type(8)))  _Float16 v8h;
typedef __attribute__((ext_vector_type(8)))  float    v8f;

#define B_ 1024
#define T_ 512
#define D_ 128
#define K_ 256    // 2*D

#define WS_ 264   // LDS weight row stride in halves (528B -> conflict-free banks)
#define XS_ 136   // LDS activation row stride in halves (272B -> conflict-free banks)

// ---------------------------------------------------------------------------
// Prep: convert weights to f16 and transpose to [N][K] (unpadded in global ws)
// so WMMA B-fragments become two contiguous 8-half runs per lane.
// ---------------------------------------------------------------------------
__global__ void augru_prep(const float* __restrict__ gateK,
                           const float* __restrict__ candK,
                           half_t* __restrict__ gwt,
                           half_t* __restrict__ cwt)
{
    int idx = blockIdx.x * blockDim.x + threadIdx.x;
    if (idx < K_ * K_) {
        int k = idx >> 8, n = idx & 255;
        gwt[n * K_ + k] = (half_t)gateK[idx];
    } else {
        int e = idx - K_ * K_;
        if (e < K_ * D_) {
            int k = e >> 7, n = e & 127;
            cwt[n * K_ + k] = (half_t)candK[e];
        }
    }
}

__device__ __forceinline__ float sigmoidf_(float x) {
    return 1.0f / (1.0f + __expf(-x));
}

// A fragment (16x32 f16): lane<16 -> row M=lane,   K = {k0..k0+7,  k0+16..k0+23}
//                         lane>=16 -> row M=lane-16, K = {k0+8..15, k0+24..31}
__device__ __forceinline__ v16h load_a_frag(const half_t* rowbase, int hi8) {
    v8h a0 = *(const v8h*)(rowbase + hi8);
    v8h a1 = *(const v8h*)(rowbase + 16 + hi8);
    v16h r;
#pragma unroll
    for (int i = 0; i < 8; ++i) { r[i] = a0[i]; r[i + 8] = a1[i]; }
    return r;
}

// B fragment (32x16 f16), LDS weights stored [N][WS_]:
// lane<16: N=n_base+lane, K={k0..k0+7, k0+16..k0+23} ; lane>=16: N=n_base+lane-16, K={k0+8..15, k0+24..31}
__device__ __forceinline__ v16h load_b_frag(const half_t* w, int lane, int n_base, int k0) {
    int n  = n_base + (lane & 15);
    int kb = k0 + ((lane >> 4) << 3);
    const v8h* p = (const v8h*)(w + n * WS_ + kb);
    v8h b0 = p[0], b1 = p[2];
    v16h r;
#pragma unroll
    for (int i = 0; i < 8; ++i) { r[i] = b0[i]; r[i + 8] = b1[i]; }
    return r;
}

// ---------------------------------------------------------------------------
// Main: one block = 16 batch rows, 8 wave32, full T=512 recurrence.
// Bank-conflict-free padded LDS, register-pipelined x/att prefetch,
// 2 barriers per step (double-buffered x tile + att staging).
// ---------------------------------------------------------------------------
__global__ void __launch_bounds__(256) augru_main(
    const float* __restrict__ x,        // [B,T,D]
    const float* __restrict__ att,      // [B,T,1]
    const float* __restrict__ gate_b,   // [256]
    const float* __restrict__ cand_b,   // [128]
    const int*   __restrict__ seqlen,   // [B,1]
    const half_t* __restrict__ gwt,     // [256][256] f16 [N][K]
    const half_t* __restrict__ cwt,     // [128][256] f16 [N][K]
    float* __restrict__ out)            // [B,T,D]
{
    extern __shared__ char smem_raw[];
    half_t* gw  = (half_t*)smem_raw;            // 256 * WS_
    half_t* cw  = gw + 256 * WS_;               // 128 * WS_
    half_t* xs  = cw + 128 * WS_;               // 2 * 16 * XS_ (double buffer)
    half_t* hs  = xs + 2 * 16 * XS_;            // 16 * XS_
    half_t* rh  = hs + 16 * XS_;                // 16 * XS_
    float*  us  = (float*)(rh + 16 * XS_);      // 16 * XS_ f32
    float*  as_ = us + 16 * XS_;                // 2 * 16 (double buffer)

    const int tid  = threadIdx.x;
    const int wave = tid >> 5;
    const int lane = tid & 31;
    const int b0   = blockIdx.x << 4;

    // ---- Stage weights into padded LDS rows (16B vectors), init h = 0 ----
    {
        const int4* s0 = (const int4*)gwt;
        for (int i = tid; i < 256 * 32; i += 256) {
            int r = i >> 5, c = i & 31;
            *(int4*)(gw + r * WS_ + c * 8) = s0[r * 32 + c];
        }
        const int4* s1 = (const int4*)cwt;
        for (int i = tid; i < 128 * 32; i += 256) {
            int r = i >> 5, c = i & 31;
            *(int4*)(cw + r * WS_ + c * 8) = s1[r * 32 + c];
        }
    }
    for (int i = tid; i < (16 * XS_) / 2; i += 256) ((unsigned int*)hs)[i] = 0u;

    const int M   = lane & 15;
    const int hi8 = (lane >> 4) << 3;           // 0 or 8

    // Per-thread loop-invariant values
    const int gc0 = wave * 32 + (lane & 15);
    const int gc1 = gc0 + 16;
    const float gb0 = gate_b[gc0];
    const float gb1 = gate_b[gc1];
    const int nc  = wave * 16 + (lane & 15);
    const float cb = cand_b[nc];
    const bool r_wave = (wave < 4);
    const int uc0 = gc0 - 128;
    const int uc1 = gc1 - 128;

    int lenj[8];
#pragma unroll
    for (int j = 0; j < 8; ++j) lenj[j] = seqlen[b0 + j + hi8];

    // ---- Prologue: prefetch x(0), att(0) into registers ----
    float4 xp[2];
    float  ap = 0.0f;
#pragma unroll
    for (int c = 0; c < 2; ++c) {
        int e = tid + 256 * c;
        int r = e >> 5, d4 = (e & 31) << 2;
        xp[c] = *(const float4*)(x + ((size_t)(b0 + r) * T_) * D_ + d4);
    }
    if (tid < 16) ap = att[(size_t)(b0 + tid) * T_];

    for (int t = 0; t < T_; ++t) {
        half_t* xsb = xs  + (t & 1) * (16 * XS_);
        float*  asb = as_ + (t & 1) * 16;

        // ---- stage x(t) (f32 regs -> f16 LDS) and att(t) ----
#pragma unroll
        for (int c = 0; c < 2; ++c) {
            int e = tid + 256 * c;
            int r = e >> 5, d4 = (e & 31) << 2;
            half_t* dp = xsb + r * XS_ + d4;
            dp[0] = (half_t)xp[c].x; dp[1] = (half_t)xp[c].y;
            dp[2] = (half_t)xp[c].z; dp[3] = (half_t)xp[c].w;
        }
        if (tid < 16) asb[tid] = ap;

        // ---- issue prefetch of x(t+1)/att(t+1); overlaps full step compute ----
        if (t + 1 < T_) {
#pragma unroll
            for (int c = 0; c < 2; ++c) {
                int e = tid + 256 * c;
                int r = e >> 5, d4 = (e & 31) << 2;
                xp[c] = *(const float4*)(x + ((size_t)(b0 + r) * T_ + (t + 1)) * D_ + d4);
            }
            if (tid < 16) ap = att[(size_t)(b0 + tid) * T_ + (t + 1)];
        }
        __syncthreads();   // barrier 1: x staged, h(t-1) final

        // ---- gate GEMM: gi = [x,h] @ Wg ; this wave owns 2 N-tiles ----
        {
            v8f acc0 = {}; v8f acc1 = {};
#pragma unroll
            for (int ki = 0; ki < 8; ++ki) {
                const half_t* arow = (ki < 4) ? (xsb + M * XS_ + ki * 32)
                                              : (hs  + M * XS_ + (ki - 4) * 32);
                v16h a   = load_a_frag(arow, hi8);
                v16h bv0 = load_b_frag(gw, lane, wave * 32,      ki * 32);
                v16h bv1 = load_b_frag(gw, lane, wave * 32 + 16, ki * 32);
                acc0 = __builtin_amdgcn_wmma_f32_16x16x32_f16(
                        false, a, false, bv0, (short)0, acc0, false, false);
                acc1 = __builtin_amdgcn_wmma_f32_16x16x32_f16(
                        false, a, false, bv1, (short)0, acc1, false, false);
            }
            if (r_wave) {   // waves 0-3: r = sigmoid(.); rh = r*h
#pragma unroll
                for (int j = 0; j < 8; ++j) {
                    int Mr = j + hi8;
                    float r0 = sigmoidf_(acc0[j] + gb0);
                    float r1 = sigmoidf_(acc1[j] + gb1);
                    rh[Mr * XS_ + gc0] = (half_t)(r0 * (float)hs[Mr * XS_ + gc0]);
                    rh[Mr * XS_ + gc1] = (half_t)(r1 * (float)hs[Mr * XS_ + gc1]);
                }
            } else {        // waves 4-7: u = sigmoid(.)
#pragma unroll
                for (int j = 0; j < 8; ++j) {
                    int Mr = j + hi8;
                    us[Mr * XS_ + uc0] = sigmoidf_(acc0[j] + gb0);
                    us[Mr * XS_ + uc1] = sigmoidf_(acc1[j] + gb1);
                }
            }
        }
        __syncthreads();   // barrier 2: rh/us ready

        // ---- cand GEMM: c = tanh([x, r*h] @ Wc); combine; update h; write out ----
        {
            v8f acc = {};
#pragma unroll
            for (int ki = 0; ki < 8; ++ki) {
                const half_t* arow = (ki < 4) ? (xsb + M * XS_ + ki * 32)
                                              : (rh  + M * XS_ + (ki - 4) * 32);
                v16h a  = load_a_frag(arow, hi8);
                v16h bv = load_b_frag(cw, lane, wave * 16, ki * 32);
                acc = __builtin_amdgcn_wmma_f32_16x16x32_f16(
                        false, a, false, bv, (short)0, acc, false, false);
            }
#pragma unroll
            for (int j = 0; j < 8; ++j) {
                int Mr = j + hi8;
                float c     = tanhf(acc[j] + cb);
                float u     = us[Mr * XS_ + nc];
                float a_att = asb[Mr];
                float uu    = (1.0f - a_att) * u;
                float hold  = (float)hs[Mr * XS_ + nc];
                float hnew  = uu * hold + (1.0f - uu) * c;
                bool  m     = (t < lenj[j]);
                float hnext = m ? hnew : hold;
                out[((size_t)(b0 + Mr) * T_ + t) * D_ + nc] = m ? hnew : 0.0f;
                hs[Mr * XS_ + nc] = (half_t)hnext;
            }
        }
        // no end-of-step barrier: next iteration's barrier 1 covers h/rh/us
        // hazards; x/att staging is double-buffered.
    }
}

extern "C" void kernel_launch(void* const* d_in, const int* in_sizes, int n_in,
                              void* d_out, int out_size, void* d_ws, size_t ws_size,
                              hipStream_t stream) {
    const float* x     = (const float*)d_in[0];   // rnn_input   [B,T,D]
    const float* att   = (const float*)d_in[1];   // att_score   [B,T,1]
    const float* gateK = (const float*)d_in[2];   // gate_kernel [256,256]
    const float* gbias = (const float*)d_in[3];   // gate_bias   [256]
    const float* candK = (const float*)d_in[4];   // cand_kernel [256,128]
    const float* cbias = (const float*)d_in[5];   // cand_bias   [128]
    const int*   slen  = (const int*)d_in[6];     // sequence_length [B,1]
    float* out = (float*)d_out;

    half_t* gwt = (half_t*)d_ws;          // 256*256 f16 (unpadded, [N][K])
    half_t* cwt = gwt + K_ * K_;          // 128*256 f16

    int prep_elems = K_ * K_ + K_ * D_;
    augru_prep<<<(prep_elems + 255) / 256, 256, 0, stream>>>(gateK, candK, gwt, cwt);

    size_t smem = (size_t)(256 * WS_ + 128 * WS_          // padded weights
                         + 2 * 16 * XS_                   // x double buffer
                         + 16 * XS_ + 16 * XS_) * sizeof(half_t)   // h, r*h
                + (size_t)(16 * XS_) * sizeof(float)      // u
                + (size_t)(2 * 16) * sizeof(float);       // att double buffer
                                                          // = 228,992 B < 320KB

    augru_main<<<B_ / 16, 256, smem, stream>>>(x, att, gbias, cbias, slen, gwt, cwt, out);
}
